// CausalSelfAttention_43490838840079
// MI455X (gfx1250) — compile-verified, matrix operand order checked by reading, not run on hardware
//
#include <hip/hip_runtime.h>

typedef unsigned short u16;
typedef __bf16 bf16_t;
typedef bf16_t v16bf __attribute__((ext_vector_type(16)));
typedef float  v8f   __attribute__((ext_vector_type(8)));

// ---------------- scalar conversion helpers ----------------
__device__ __forceinline__ u16 f2bf(float x){
    union { float f; unsigned u; } c; c.f = x;
    unsigned u = c.u;
    u += 0x7fffu + ((u >> 16) & 1u);   // round-to-nearest-even
    return (u16)(u >> 16);
}
__device__ __forceinline__ float bf2f(u16 x){
    union { unsigned u; float f; } c; c.u = ((unsigned)x) << 16;
    return c.f;
}
__device__ __forceinline__ unsigned pack2bf(float lo, float hi){
    return ((unsigned)f2bf(hi) << 16) | (unsigned)f2bf(lo);
}
__device__ __forceinline__ void store_c(float* p, float v){ *p = v; }
__device__ __forceinline__ void store_c(u16* p, float v){ *p = f2bf(v); }

// load 4 consecutive elements, return them as 4 packed bf16 (uint2)
__device__ __forceinline__ uint2 load4_bf16(const float* p){
    float4 f = *reinterpret_cast<const float4*>(p);
    uint2 r; r.x = pack2bf(f.x, f.y); r.y = pack2bf(f.z, f.w);
    return r;
}
__device__ __forceinline__ uint2 load4_bf16(const u16* p){
    return *reinterpret_cast<const uint2*>(p);   // already bf16
}

// ---------------- DPP 16-lane ring reductions (VALU, no LDS) ----------------
// row_ror:N dpp_ctrl = 0x120|N ; rotation tree-reduce over the 16-lane DPP row
// is exact for commutative ops, and 16-lane rows never mix the two wave halves.
template<int CTRL>
__device__ __forceinline__ float dpp_rot(float x){
    int xi = __builtin_bit_cast(int, x);
    int r  = __builtin_amdgcn_update_dpp(0, xi, CTRL, 0xf, 0xf, true);
    return __builtin_bit_cast(float, r);
}
__device__ __forceinline__ float row_max16(float v){
    v = fmaxf(v, dpp_rot<0x121>(v));
    v = fmaxf(v, dpp_rot<0x122>(v));
    v = fmaxf(v, dpp_rot<0x124>(v));
    v = fmaxf(v, dpp_rot<0x128>(v));
    return v;
}
__device__ __forceinline__ float row_sum16(float v){
    v += dpp_rot<0x121>(v);
    v += dpp_rot<0x122>(v);
    v += dpp_rot<0x124>(v);
    v += dpp_rot<0x128>(v);
    return v;
}

// ---------------- WMMA fragment helpers ----------------
union Frag {
    v16bf v;
    u16   s[16];
    uint4 q[2];
};

// Row fragment (A layout; also B when the tile is stored transposed [n][k]).
__device__ __forceinline__ v16bf lds_frag_row(const u16* p, int rowBase, int colBase,
                                              int stride, int lane){
    int r    = lane & 15;
    int koff = (lane >> 4) << 3;
    const u16* row = p + (rowBase + r) * stride + colBase;
    Frag f;
    f.q[0] = *reinterpret_cast<const uint4*>(row + koff);
    f.q[1] = *reinterpret_cast<const uint4*>(row + 16 + koff);
    return f.v;
}

// Column fragment from a row-major [k][n] LDS tile (used for V in PV matmul).
__device__ __forceinline__ v16bf lds_frag_col(const u16* p, int kBase, int nBase,
                                              int stride, int lane){
    int n    = lane & 15;
    int koff = (lane >> 4) << 3;
    Frag f;
#pragma unroll
    for (int e = 0; e < 8; ++e){
        f.s[e]     = p[(kBase + koff + e)      * stride + nBase + n];
        f.s[8 + e] = p[(kBase + 16 + koff + e) * stride + nBase + n];
    }
    return f.v;
}

// ---------------- generic bf16 WMMA GEMM:  C(MxN) = A(MxK) @ B(KxN) ----------------
// Block tile 128x128, K-step 32, 256 threads (8 waves), each wave -> 32x64 (2x4 frags).
template<typename TA, typename TC>
__global__ __launch_bounds__(256)
void gemm_wmma_kernel(const TA* __restrict__ A, const float* __restrict__ B,
                      TC* __restrict__ C, int M, int N, int K)
{
    __shared__ __align__(16) u16 As[128 * 32];
    __shared__ __align__(16) u16 Bs[128 * 32];   // transposed: [n][k]

    const int tid  = threadIdx.x;
    const int lane = tid & 31;
    const int wave = tid >> 5;
    const int wm   = wave >> 1;     // 0..3
    const int wn   = wave & 1;      // 0..1
    const int bm   = blockIdx.y * 128;
    const int bn   = blockIdx.x * 128;

    // A-staging group of this thread (4 float4 groups of the 128x32 tile)
    const int ar0 = tid >> 3, ac0 = (tid & 7) * 4;       // group i stride: +32 rows
    // B-staging group (4 float4 groups of the 32x128 tile)
    const int br0 = tid >> 5, bc0 = (tid & 31) * 4;      // group i stride: +8 k-rows

    v8f acc[2][4];
#pragma unroll
    for (int i = 0; i < 2; ++i)
#pragma unroll
        for (int j = 0; j < 4; ++j) acc[i][j] = {};

    for (int k0 = 0; k0 < K; k0 += 32){
        // prefetch next K-chunk (global_prefetch_b8 path)
        if (k0 + 32 < K){
            __builtin_prefetch(&A[(size_t)(bm + ar0) * K + k0 + 32 + ac0], 0, 0);
            __builtin_prefetch(&B[(size_t)(k0 + 32 + br0) * N + bn + bc0], 0, 0);
        }
        // stage A tile (128x32) into LDS as bf16: b128 loads, b64 LDS stores
#pragma unroll
        for (int i = 0; i < 4; ++i){
            int r = ar0 + i * 32;
            uint2 v = load4_bf16(&A[(size_t)(bm + r) * K + k0 + ac0]);
            *reinterpret_cast<uint2*>(&As[r * 32 + ac0]) = v;
        }
        // stage B tile (32x128) transposed into LDS as bf16
#pragma unroll
        for (int i = 0; i < 4; ++i){
            int r = br0 + i * 8;
            float4 f = *reinterpret_cast<const float4*>(&B[(size_t)(k0 + r) * N + bn + bc0]);
            Bs[(bc0 + 0) * 32 + r] = f2bf(f.x);
            Bs[(bc0 + 1) * 32 + r] = f2bf(f.y);
            Bs[(bc0 + 2) * 32 + r] = f2bf(f.z);
            Bs[(bc0 + 3) * 32 + r] = f2bf(f.w);
        }
        __syncthreads();

        v16bf a[2], b[4];
#pragma unroll
        for (int fm = 0; fm < 2; ++fm)
            a[fm] = lds_frag_row(As, wm * 32 + fm * 16, 0, 32, lane);
#pragma unroll
        for (int fn = 0; fn < 4; ++fn)
            b[fn] = lds_frag_row(Bs, wn * 64 + fn * 16, 0, 32, lane);
#pragma unroll
        for (int fm = 0; fm < 2; ++fm)
#pragma unroll
            for (int fn = 0; fn < 4; ++fn)
                acc[fm][fn] = __builtin_amdgcn_wmma_f32_16x16x32_bf16(
                    false, a[fm], false, b[fn], (short)0, acc[fm][fn], false, false);
        __syncthreads();
    }

    const int hal = lane >> 4, c16 = lane & 15;
#pragma unroll
    for (int fm = 0; fm < 2; ++fm)
#pragma unroll
        for (int fn = 0; fn < 4; ++fn)
#pragma unroll
            for (int i = 0; i < 8; ++i){
                int r = bm + wm * 32 + fm * 16 + i + 8 * hal;
                int c = bn + wn * 64 + fn * 16 + c16;
                store_c(&C[(size_t)r * N + c], acc[fm][fn][i]);
            }
}

// ---------------- RoPE + QKV scatter ----------------
__global__ __launch_bounds__(256)
void rope_kernel(const u16* __restrict__ qkv, u16* __restrict__ Q,
                 u16* __restrict__ Kd, u16* __restrict__ V)
{
    int t = blockIdx.x * 256 + threadIdx.x;      // 22-bit index space
    int j = t & 31;
    int h = (t >> 5) & 15;
    int s = (t >> 9) & 2047;
    int b = t >> 20;

    size_t src = ((size_t)(b * 2048 + s)) * 3072 + h * 64 + j;
    size_t dst = ((size_t)((b * 16 + h) * 2048 + s)) * 64 + j;

    float inv_freq = expf(-(float)j * (9.210340371976184f / 32.0f)); // 10000^(-j/32)
    float th = (float)s * inv_freq;
    float c = cosf(th), sn = sinf(th);

    float q0 = bf2f(qkv[src]),        q1 = bf2f(qkv[src + 32]);
    Q[dst]      = f2bf(q0 * c - q1 * sn);
    Q[dst + 32] = f2bf(q1 * c + q0 * sn);

    float k0 = bf2f(qkv[src + 1024]), k1 = bf2f(qkv[src + 1024 + 32]);
    Kd[dst]      = f2bf(k0 * c - k1 * sn);
    Kd[dst + 32] = f2bf(k1 * c + k0 * sn);

    V[dst]      = qkv[src + 2048];
    V[dst + 32] = qkv[src + 2048 + 32];
}

// ---------------- causal flash attention ----------------
// Q,K,V: (B,H,S,64) bf16.  O: (B,S,1024) bf16.
__global__ __launch_bounds__(128)
void attn_kernel(const u16* __restrict__ Q, const u16* __restrict__ Kg,
                 const u16* __restrict__ Vg, u16* __restrict__ O)
{
    __shared__ __align__(16) u16 Ks[64 * 64];
    __shared__ __align__(16) u16 Vs[64 * 64];
    __shared__ __align__(16) u16 Ps[4][16 * 32];   // per-wave P staging

    const int qt   = blockIdx.x;          // 0..31 (64-row q tiles)
    const int h    = blockIdx.y;
    const int b    = blockIdx.z;
    const int tid  = threadIdx.x;
    const int lane = tid & 31;
    const int wave = tid >> 5;
    const int qbase = qt * 64 + wave * 16;
    // scalar (SGPR) copy of the wave-uniform q strip base: keeps the causal
    // skip as an SCC branch instead of EXEC-divergent flow around WMMA.
    const int qbS  = __builtin_amdgcn_readfirstlane(qbase);
    const int hal  = lane >> 4, c16 = lane & 15;

    const size_t headBase = (size_t)((b * 16 + h) * 2048) * 64;

    // load this wave's Q strip (16x64) as two A-fragments, directly from global
    v16bf qa[2];
    {
        int r = lane & 15;
        int koff = (lane >> 4) << 3;
        const u16* row = Q + headBase + (size_t)(qbase + r) * 64;
        Frag f0, f1;
        f0.q[0] = *reinterpret_cast<const uint4*>(row + koff);
        f0.q[1] = *reinterpret_cast<const uint4*>(row + 16 + koff);
        f1.q[0] = *reinterpret_cast<const uint4*>(row + 32 + koff);
        f1.q[1] = *reinterpret_cast<const uint4*>(row + 48 + koff);
        qa[0] = f0.v; qa[1] = f1.v;
    }

    v8f o[4]; o[0] = {}; o[1] = {}; o[2] = {}; o[3] = {};
    float m[8], l[8];
#pragma unroll
    for (int i = 0; i < 8; ++i){ m[i] = -1e30f; l[i] = 0.0f; }

    u16* pl = &Ps[wave][0];

    for (int kt = 0; kt <= qt; ++kt){
        __syncthreads();
        // stage 64 keys of K and V (64x64 bf16 each) into LDS via b128 copies
        const uint4* ks4 = reinterpret_cast<const uint4*>(Kg + headBase + (size_t)(kt * 64) * 64);
        const uint4* vs4 = reinterpret_cast<const uint4*>(Vg + headBase + (size_t)(kt * 64) * 64);
        uint4* kd4 = reinterpret_cast<uint4*>(Ks);
        uint4* vd4 = reinterpret_cast<uint4*>(Vs);
#pragma unroll
        for (int i = 0; i < 4; ++i){
            int idx = tid + i * 128;
            kd4[idx] = ks4[idx];
            vd4[idx] = vs4[idx];
        }
        __syncthreads();

        for (int g = 0; g < 2; ++g){
            int kb = kt * 64 + g * 32;            // first key of this 32-key group
            if (kb > qbS + 15) break;             // scalar, wave-uniform

            // scores: two 16-key tiles, each = 2 WMMAs over hd
            v8f s0 = {}, s1 = {};
            v16bf kb0lo = lds_frag_row(Ks, g * 32,      0,  64, lane);
            v16bf kb0hi = lds_frag_row(Ks, g * 32,      32, 64, lane);
            v16bf kb1lo = lds_frag_row(Ks, g * 32 + 16, 0,  64, lane);
            v16bf kb1hi = lds_frag_row(Ks, g * 32 + 16, 32, 64, lane);
            s0 = __builtin_amdgcn_wmma_f32_16x16x32_bf16(false, qa[0], false, kb0lo, (short)0, s0, false, false);
            s0 = __builtin_amdgcn_wmma_f32_16x16x32_bf16(false, qa[1], false, kb0hi, (short)0, s0, false, false);
            s1 = __builtin_amdgcn_wmma_f32_16x16x32_bf16(false, qa[0], false, kb1lo, (short)0, s1, false, false);
            s1 = __builtin_amdgcn_wmma_f32_16x16x32_bf16(false, qa[1], false, kb1hi, (short)0, s1, false, false);

            const bool needmask = (kb + 31 > qbS); // scalar
            float p0[8], p1[8];
#pragma unroll
            for (int i = 0; i < 8; ++i){
                float v0 = s0[i] * 0.125f;        // 1/sqrt(64)
                float v1 = s1[i] * 0.125f;
                int row = qbase + i + 8 * hal;
                if (needmask){
                    if (kb + c16 > row)      v0 = -1e30f;
                    if (kb + 16 + c16 > row) v1 = -1e30f;
                }
                float mx = row_max16(fmaxf(v0, v1));      // DPP, no LDS
                float nm = fmaxf(m[i], mx);
                float sc = __expf(m[i] - nm);
                v0 = __expf(v0 - nm);
                v1 = __expf(v1 - nm);
                float rs = row_sum16(v0 + v1);            // DPP, no LDS
                l[i] = l[i] * sc + rs;
                m[i] = nm;
                o[0][i] *= sc; o[1][i] *= sc; o[2][i] *= sc; o[3][i] *= sc;
                p0[i] = v0; p1[i] = v1;
            }

            // restage P (16x32) through LDS to become an A-fragment
#pragma unroll
            for (int i = 0; i < 8; ++i){
                int row = i + 8 * hal;
                pl[row * 32 + c16]      = f2bf(p0[i]);
                pl[row * 32 + 16 + c16] = f2bf(p1[i]);
            }
            v16bf pa = lds_frag_row(pl, 0, 0, 32, lane);

            // O += P(16x32) @ V(32x64): 4 WMMAs over hd columns
#pragma unroll
            for (int n = 0; n < 4; ++n){
                v16bf vb = lds_frag_col(Vs, g * 32, n * 16, 64, lane);
                o[n] = __builtin_amdgcn_wmma_f32_16x16x32_bf16(
                    false, pa, false, vb, (short)0, o[n], false, false);
            }
        }
    }

    // normalize and scatter to (B, S, H*64) bf16
#pragma unroll
    for (int i = 0; i < 8; ++i){
        int srow = qbase + i + 8 * hal;
        float il = 1.0f / l[i];
        size_t base = ((size_t)(b * 2048 + srow)) * 1024 + h * 64;
#pragma unroll
        for (int n = 0; n < 4; ++n)
            O[base + n * 16 + c16] = f2bf(o[n][i] * il);
    }
}

// ---------------- host launcher ----------------
extern "C" void kernel_launch(void* const* d_in, const int* in_sizes, int n_in,
                              void* d_out, int out_size, void* d_ws, size_t ws_size,
                              hipStream_t stream)
{
    (void)in_sizes; (void)n_in; (void)out_size; (void)ws_size;
    const int B = 4, S = 2048, D = 1024, H = 16;
    const int BS = B * S;                     // 8192

    const float* x    = (const float*)d_in[0];   // (B*S, D)
    const float* Wqkv = (const float*)d_in[1];   // (D, 3D)
    const float* Wout = (const float*)d_in[2];   // (D, D)
    // d_in[3] = mask: causality computed analytically, not read
    float* out = (float*)d_out;                  // (B*S, D)

    u16* qkv = (u16*)d_ws;                                   // BS*3D
    u16* q   = qkv + (size_t)BS * 3 * D;                     // B*H*S*64
    u16* k   = q   + (size_t)B * H * S * 64;
    u16* v   = k   + (size_t)B * H * S * 64;
    u16* ao  = v   + (size_t)B * H * S * 64;                 // BS*D bf16

    dim3 g1(3 * D / 128, BS / 128);                          // (24, 64)
    gemm_wmma_kernel<float, u16><<<g1, 256, 0, stream>>>(x, Wqkv, qkv, BS, 3 * D, D);

    rope_kernel<<<(B * S * H * 32) / 256, 256, 0, stream>>>(qkv, q, k, v);

    dim3 g3(S / 64, H, B);                                   // (32, 16, 4)
    attn_kernel<<<g3, 128, 0, stream>>>(q, k, v, ao);

    dim3 g4(D / 128, BS / 128);                              // (8, 64)
    gemm_wmma_kernel<u16, float><<<g4, 256, 0, stream>>>(ao, Wout, out, BS, D, D);
}
